// CorrNet_20323785245274
// MI455X (gfx1250) — matrix operand order repaired and checked by reading, not compiled, not run on hardware
//
#include <hip/hip_runtime.h>
#include <hip/hip_bf16.h>
#include <cstdint>

#define N_POINTS 20000
#define NTILE    1250          // 20000 / 16
#define LEAK     0.2f
#define EPS_GN   1e-5f
#define EPS_NORM 1e-12f

typedef __attribute__((ext_vector_type(16))) _Float16 v16h;
typedef __attribute__((ext_vector_type(8)))  float    v8f;

union HU { v16h h; unsigned int u[8]; };

// ---------------------------------------------------------------------------
// helpers
// ---------------------------------------------------------------------------
__device__ __forceinline__ float lrelu(float a) { return a > 0.f ? a : LEAK * a; }

// monotone int key for float max (works for negatives), for LDS atomicMax
__device__ __forceinline__ int fkey(float f) {
    int i = __float_as_int(f);
    return (i < 0) ? (i ^ 0x7fffffff) : i;
}
__device__ __forceinline__ float fdecode(int k) {
    int i = (k < 0) ? (k ^ 0x7fffffff) : k;
    return __int_as_float(i);
}

// One "block": y = GroupNorm(leaky_relu(x @ W + b)); every group is 32 channels.
template <int K, int C>
__device__ inline void linblock(const float* x,
                                const float* __restrict__ W,
                                const float* __restrict__ b,
                                const float* __restrict__ gw,
                                const float* __restrict__ gb,
                                float* y) {
    for (int c = 0; c < C; ++c) {
        float a = b[c];
        for (int k = 0; k < K; ++k) a += x[k] * W[k * C + c];
        y[c] = lrelu(a);
    }
    for (int g = 0; g < C; g += 32) {
        float mu = 0.f;
        for (int j = 0; j < 32; ++j) mu += y[g + j];
        mu *= (1.f / 32.f);
        float var = 0.f;
        for (int j = 0; j < 32; ++j) { float d = y[g + j] - mu; var += d * d; }
        var *= (1.f / 32.f);
        float inv = rsqrtf(var + EPS_GN);
        for (int j = 0; j < 32; ++j)
            y[g + j] = (y[g + j] - mu) * inv * gw[g + j] + gb[g + j];
    }
}

// ---------------------------------------------------------------------------
// K1: mlp1 (3 blocks) -> f[128] stored; mlp2 block -> per-block channel maxes
// ---------------------------------------------------------------------------
__global__ __launch_bounds__(256) void pn_part1_kernel(
    const float* __restrict__ x, int N,
    const float* W1, const float* b1, const float* g1w, const float* g1b,   // 3->32   g=1
    const float* W2, const float* b2, const float* g2w, const float* g2b,   // 32->64  g=2
    const float* W3, const float* b3, const float* g3w, const float* g3b,   // 64->128 g=4
    const float* W4, const float* b4, const float* g4w, const float* g4b,   // 128->128 g=4 (mlp2)
    float* __restrict__ f_out, float* __restrict__ partial) {
    __shared__ int sh[128];
    const int tid = threadIdx.x;
    for (int c = tid; c < 128; c += blockDim.x) sh[c] = INT_MIN;
    __syncthreads();

    const int t = blockIdx.x * 256 + tid;
    if (t < N) {
        float xin[3];
        xin[0] = x[t * 3 + 0]; xin[1] = x[t * 3 + 1]; xin[2] = x[t * 3 + 2];
        float h1[32];  linblock<3, 32>(xin, W1, b1, g1w, g1b, h1);
        float h2[64];  linblock<32, 64>(h1, W2, b2, g2w, g2b, h2);
        float f[128];  linblock<64, 128>(h2, W3, b3, g3w, g3b, f);
        for (int c = 0; c < 128; ++c) f_out[(size_t)t * 128 + c] = f[c];

        // mlp2 computed in 32-channel chunks (== its groupnorm groups)
        for (int g = 0; g < 4; ++g) {
            float ch[32];
            for (int j = 0; j < 32; ++j) {
                const int c = g * 32 + j;
                float a = b4[c];
                for (int k = 0; k < 128; ++k) a += f[k] * W4[k * 128 + c];
                ch[j] = lrelu(a);
            }
            float mu = 0.f;
            for (int j = 0; j < 32; ++j) mu += ch[j];
            mu *= (1.f / 32.f);
            float var = 0.f;
            for (int j = 0; j < 32; ++j) { float d = ch[j] - mu; var += d * d; }
            var *= (1.f / 32.f);
            float inv = rsqrtf(var + EPS_GN);
            for (int j = 0; j < 32; ++j) {
                float v = (ch[j] - mu) * inv * g4w[g * 32 + j] + g4b[g * 32 + j];
                atomicMax(&sh[g * 32 + j], fkey(v));
            }
        }
    }
    __syncthreads();
    for (int c = tid; c < 128; c += blockDim.x)
        partial[(size_t)blockIdx.x * 128 + c] = fdecode(sh[c]);
}

// ---------------------------------------------------------------------------
// K2: reduce per-block partial maxes -> gmax[128]
// ---------------------------------------------------------------------------
__global__ void reduce_max_kernel(const float* __restrict__ partial, int nb,
                                  float* __restrict__ gmax) {
    const int c = threadIdx.x;
    if (c >= 128) return;
    float m = -3.4e38f;
    for (int b = 0; b < nb; ++b) m = fmaxf(m, partial[(size_t)b * 128 + c]);
    gmax[c] = m;
}

// ---------------------------------------------------------------------------
// K3: concat(f, gmax) -> mlp3 -> lin -> l2norm; write f32 features + f16 copy
// ---------------------------------------------------------------------------
__global__ __launch_bounds__(256) void pn_part2_kernel(
    const float* __restrict__ f_in, const float* __restrict__ gmax, int N,
    const float* W5, const float* b5, const float* g5w, const float* g5b,   // 256->128 g=4
    const float* W6, const float* b6, const float* g6w, const float* g6b,   // 128->64  g=2
    const float* Wl, const float* bl,                                       // 64->128
    float* __restrict__ norm_out, _Float16* __restrict__ h16_out) {
    __shared__ float sg[128];
    for (int c = threadIdx.x; c < 128; c += blockDim.x) sg[c] = gmax[c];
    __syncthreads();

    const int t = blockIdx.x * 256 + threadIdx.x;
    if (t >= N) return;

    float cat[256];
    for (int c = 0; c < 128; ++c) cat[c] = f_in[(size_t)t * 128 + c];
    for (int c = 0; c < 128; ++c) cat[128 + c] = sg[c];

    float z1[128]; linblock<256, 128>(cat, W5, b5, g5w, g5b, z1);
    float z2[64];  linblock<128, 64>(z1, W6, b6, g6w, g6b, z2);

    float feat[128];
    float ss = 0.f;
    for (int c = 0; c < 128; ++c) {
        float a = bl[c];
        for (int k = 0; k < 64; ++k) a += z2[k] * Wl[k * 128 + c];
        feat[c] = a;
        ss += a * a;
    }
    const float inv = 1.f / fmaxf(sqrtf(ss), EPS_NORM);
    for (int c = 0; c < 128; ++c) {
        const float v = feat[c] * inv;
        norm_out[(size_t)t * 128 + c] = v;
        h16_out[(size_t)t * 128 + c] = (_Float16)v;
    }
}

// ---------------------------------------------------------------------------
// K4: sim = A @ B^T via V_WMMA_F32_16X16X32_F16, fused row max/argmax.
// One wave per 16-row strip; A fragments resident in VGPRs for whole sweep;
// B tile stream double-buffered in VGPRs (loads of tile ct+1 fly behind the
// four WMMAs of tile ct).
// ---------------------------------------------------------------------------
__device__ __forceinline__ void load_btile(const _Float16* __restrict__ B16,
                                           int ct, int l15, int half, v16h* b) {
    // B fragment layout: lanes 0-15 hold K=0..15, lanes 16-31 K=16..31 (per
    // half), column n = lane%16 -> 8 consecutive dwords of the B row.
    const unsigned int* brow =
        (const unsigned int*)(B16 + (size_t)(ct * 16 + l15) * 128);
#pragma unroll
    for (int kt = 0; kt < 4; ++kt) {
        HU u;
#pragma unroll
        for (int v = 0; v < 8; ++v) u.u[v] = brow[kt * 16 + 8 * half + v];
        b[kt] = u.h;
    }
}

__device__ __forceinline__ void tile_mma_update(const v16h* a, const v16h* b,
                                                int col, float* rmax, int* ridx) {
    v8f acc = {};
#pragma unroll
    for (int kt = 0; kt < 4; ++kt)
        acc = __builtin_amdgcn_wmma_f32_16x16x32_f16(
            false, a[kt], false, b[kt], (short)0, acc, false, false);
#pragma unroll
    for (int i = 0; i < 8; ++i) {
        const float v = acc[i];                 // row = i + 8*half, col = col
        if (v > rmax[i]) { rmax[i] = v; ridx[i] = col; }
    }
}

__global__ __launch_bounds__(256) void sim_argmax_kernel(
    const _Float16* __restrict__ A16,   // [N,128] f16 row-major (norm_vtx)
    const _Float16* __restrict__ B16,   // [M,128] f16 row-major (norm_pts)
    float* __restrict__ rowmax, int* __restrict__ rowidx) {
    __shared__ float smax[8][16][17];
    __shared__ int   sidx[8][16][17];

    const int lane = threadIdx.x & 31;
    const int wv   = threadIdx.x >> 5;
    int strip = blockIdx.x * 8 + wv;
    if (strip > NTILE - 1) strip = NTILE - 1;   // dup work, deterministic; keeps EXEC full
    const int half = lane >> 4;
    const int l15  = lane & 15;

    // A fragments: 16x32, M=lane%16; lanes 0-15 hold K=0..15, lanes 16-31 K=16..31 (matches B).
    const unsigned int* arow =
        (const unsigned int*)(A16 + (size_t)(strip * 16 + l15) * 128);
    v16h a[4];
#pragma unroll
    for (int kt = 0; kt < 4; ++kt) {
        HU ua;
#pragma unroll
        for (int p = 0; p < 4; ++p) ua.u[p]     = arow[kt * 16 + 8 * half + p];
#pragma unroll
        for (int p = 0; p < 4; ++p) ua.u[4 + p] = arow[kt * 16 + 8 * half + 4 + p];
        a[kt] = ua.h;
    }

    float rmax[8]; int ridx[8];
#pragma unroll
    for (int i = 0; i < 8; ++i) { rmax[i] = -3.4e38f; ridx[i] = 0; }

    // Software-pipelined sweep over 1250 column tiles, 2 per iteration
    // (NTILE is even): loads for the next tile are issued before the WMMAs
    // of the current tile so s_wait_loadcnt retires behind matrix work.
    v16h b0[4], b1[4];
    load_btile(B16, 0, l15, half, b0);
    for (int ct = 0; ct < NTILE; ct += 2) {
        __builtin_prefetch(B16 + (size_t)(((ct + 4) % NTILE) * 16 + l15) * 128, 0, 3);
        load_btile(B16, ct + 1, l15, half, b1);
        tile_mma_update(a, b0, ct * 16 + l15, rmax, ridx);
        if (ct + 2 < NTILE) load_btile(B16, ct + 2, l15, half, b0);
        tile_mma_update(a, b1, (ct + 1) * 16 + l15, rmax, ridx);
    }

#pragma unroll
    for (int i = 0; i < 8; ++i) {
        const int rl = i + 8 * half;
        smax[wv][rl][l15] = rmax[i];
        sidx[wv][rl][l15] = ridx[i];
    }
    __syncthreads();

    if (lane < 16) {
        float best = -3.4e38f; int bi = 0;
        for (int c = 0; c < 16; ++c) {
            const float v = smax[wv][lane][c];
            const int  id = sidx[wv][lane][c];
            if (v > best || (v == best && id < bi)) { best = v; bi = id; }
        }
        const int row = strip * 16 + lane;
        rowmax[row] = best;
        rowidx[row] = bi;
    }
}

// ---------------------------------------------------------------------------
// K5: gather + final block (257->64, g=2) + linear (64->1)
// ---------------------------------------------------------------------------
__global__ __launch_bounds__(256) void final_mlp_kernel(
    const float* __restrict__ nv, const float* __restrict__ np,
    const float* __restrict__ pt_s, const int* __restrict__ idx, int N,
    const float* Wm, const float* bm, const float* gmw, const float* gmb,
    const float* Wl, const float* bl, float* __restrict__ out) {
    const int t = blockIdx.x * 256 + threadIdx.x;
    if (t >= N) return;

    float cat[257];
    for (int c = 0; c < 128; ++c) cat[c] = nv[(size_t)t * 128 + c];
    const size_t j = (size_t)idx[t] * 128;
    for (int c = 0; c < 128; ++c) cat[128 + c] = np[j + c];
    cat[256] = pt_s[t];

    float h[64];
    linblock<257, 64>(cat, Wm, bm, gmw, gmb, h);

    float o = bl[0];
    for (int k = 0; k < 64; ++k) o += h[k] * Wl[k];
    out[t] = o;
}

// ---------------------------------------------------------------------------
// launch
// ---------------------------------------------------------------------------
extern "C" void kernel_launch(void* const* d_in, const int* in_sizes, int n_in,
                              void* d_out, int out_size, void* d_ws, size_t ws_size,
                              hipStream_t stream) {
    const int N  = N_POINTS;
    const int NB = (N + 255) / 256;      // 79

    const float* vtx = (const float*)d_in[0];
    const float* pts = (const float*)d_in[1];
    // params flat in insertion order: pn.mlp1[0..2]{W,b,gw,gb}, pn.mlp2[0]{..},
    // pn.mlp3[0..1]{..}, pn.lin{W,b}, mlp[0]{W,b,gw,gb}, l{W,b}
    const float* P[32];
    for (int i = 0; i < 32; ++i)
        P[i] = (i + 2 < n_in) ? (const float*)d_in[i + 2] : nullptr;

    // workspace carve-out (256B aligned slices)
    char* w = (char*)d_ws;
    auto alloc = [&](size_t bytes) -> void* {
        void* p = (void*)w;
        w += (bytes + 255) & ~(size_t)255;
        return p;
    };
    float*     f_vtx  = (float*)alloc((size_t)N * 128 * sizeof(float));
    float*     f_pts  = (float*)alloc((size_t)N * 128 * sizeof(float));
    float*     part_v = (float*)alloc((size_t)NB * 128 * sizeof(float));
    float*     part_p = (float*)alloc((size_t)NB * 128 * sizeof(float));
    float*     gmax_v = (float*)alloc(128 * sizeof(float));
    float*     gmax_p = (float*)alloc(128 * sizeof(float));
    _Float16*  A16    = (_Float16*)alloc((size_t)N * 128 * sizeof(_Float16));
    _Float16*  B16    = (_Float16*)alloc((size_t)N * 128 * sizeof(_Float16));
    float*     pt_s   = (float*)alloc((size_t)N * sizeof(float));
    int*       idxb   = (int*)alloc((size_t)N * sizeof(int));

    float* out      = (float*)d_out;
    float* norm_vtx = out;                       // [N,128]
    float* norm_pts = out + (size_t)N * 128;     // [N,128]
    float* final_o  = out + (size_t)2 * N * 128; // [N,1]

    // PointNet part 1 (both clouds)
    pn_part1_kernel<<<NB, 256, 0, stream>>>(
        vtx, N,
        P[0], P[1], P[2], P[3],   P[4], P[5], P[6], P[7],
        P[8], P[9], P[10], P[11], P[12], P[13], P[14], P[15],
        f_vtx, part_v);
    pn_part1_kernel<<<NB, 256, 0, stream>>>(
        pts, N,
        P[0], P[1], P[2], P[3],   P[4], P[5], P[6], P[7],
        P[8], P[9], P[10], P[11], P[12], P[13], P[14], P[15],
        f_pts, part_p);

    reduce_max_kernel<<<1, 128, 0, stream>>>(part_v, NB, gmax_v);
    reduce_max_kernel<<<1, 128, 0, stream>>>(part_p, NB, gmax_p);

    // PointNet part 2 (features + f16 copies)
    pn_part2_kernel<<<NB, 256, 0, stream>>>(
        f_vtx, gmax_v, N,
        P[16], P[17], P[18], P[19],  P[20], P[21], P[22], P[23],
        P[24], P[25],
        norm_vtx, A16);
    pn_part2_kernel<<<NB, 256, 0, stream>>>(
        f_pts, gmax_p, N,
        P[16], P[17], P[18], P[19],  P[20], P[21], P[22], P[23],
        P[24], P[25],
        norm_pts, B16);

    // fused WMMA similarity + row max/argmax
    sim_argmax_kernel<<<(NTILE + 7) / 8, 256, 0, stream>>>(A16, B16, pt_s, idxb);

    // gather + final MLP
    final_mlp_kernel<<<NB, 256, 0, stream>>>(
        norm_vtx, norm_pts, pt_s, idxb, N,
        P[26], P[27], P[28], P[29],  P[30], P[31],
        final_o);
}
